// SchNet_28587302322453
// MI455X (gfx1250) — compile-verified, matrix operand order checked by reading, not run on hardware
//
#include <hip/hip_runtime.h>
#include <hip/hip_fp16.h>

typedef __attribute__((ext_vector_type(16))) _Float16 v16h;
typedef __attribute__((ext_vector_type(8)))  _Float16 v8h;
typedef __attribute__((ext_vector_type(8)))  float    v8f;

#define N_ATOMS  512
#define HIDDEN   128
#define FILTERS  128
#define NUM_RBF  50
#define N_MOL    16
#define CUTOFF_F 10.0f
#define N_LAYERS 3

__device__ __forceinline__ float silu_f(float v) { return v / (1.0f + __expf(-v)); }

__device__ __forceinline__ v8f wmma16(v16h a, v16h b, v8f c) {
    // D = A(16x32 f16) * B(32x16 f16) + C(16x16 f32)
    return __builtin_amdgcn_wmma_f32_16x16x32_f16(false, a, false, b, (short)0, c, false, false);
}

// Load a 16x32 A-fragment (one K-chunk) from a row-major f16 matrix with 128 cols.
// Per ISA layout: lane L (M = L%16) holds K = kc*32 + (L<16?0:8) + {0..7, 16..23}.
__device__ __forceinline__ v16h load_a_f16(const _Float16* row, int kc, int off8) {
    v8h lo = *(const v8h*)(row + kc * 32 + off8);
    v8h hi = *(const v8h*)(row + kc * 32 + off8 + 16);
    return __builtin_shufflevector(lo, hi, 0,1,2,3,4,5,6,7,8,9,10,11,12,13,14,15);
}

// Same fragment but sourced from a row-major f32 matrix (converted to f16).
__device__ __forceinline__ v16h load_a_f32(const float* row, int kc, int off8) {
    float4 f0 = *(const float4*)(row + kc * 32 + off8);
    float4 f1 = *(const float4*)(row + kc * 32 + off8 + 4);
    float4 f2 = *(const float4*)(row + kc * 32 + off8 + 16);
    float4 f3 = *(const float4*)(row + kc * 32 + off8 + 20);
    v16h a;
    a[0]=(_Float16)f0.x; a[1]=(_Float16)f0.y; a[2]=(_Float16)f0.z; a[3]=(_Float16)f0.w;
    a[4]=(_Float16)f1.x; a[5]=(_Float16)f1.y; a[6]=(_Float16)f1.z; a[7]=(_Float16)f1.w;
    a[8]=(_Float16)f2.x; a[9]=(_Float16)f2.y; a[10]=(_Float16)f2.z; a[11]=(_Float16)f2.w;
    a[12]=(_Float16)f3.x; a[13]=(_Float16)f3.y; a[14]=(_Float16)f3.z; a[15]=(_Float16)f3.w;
    return a;
}

// ---------------- prep kernels ----------------

__global__ void k_embed(const int* __restrict__ z, const float* __restrict__ emb,
                        float* __restrict__ x, _Float16* __restrict__ xf16) {
    int t = blockIdx.x * blockDim.x + threadIdx.x;            // 512*128
    int a = t >> 7, h = t & 127;
    float v = emb[z[a] * HIDDEN + h];
    x[t] = v;
    xf16[t] = (_Float16)v;
}

__global__ void k_dist_env(const float* __restrict__ pos,
                           float* __restrict__ dmat, float* __restrict__ envm) {
    int t = blockIdx.x * blockDim.x + threadIdx.x;            // 512*512
    int i = t >> 9, j = t & 511;
    float dx = pos[i*3+0] - pos[j*3+0];
    float dy = pos[i*3+1] - pos[j*3+1];
    float dz = pos[i*3+2] - pos[j*3+2];
    float d2 = dx*dx + dy*dy + dz*dz;
    float d  = (d2 > 0.0f) ? sqrtf(d2) : 0.0f;
    dmat[t] = d;
    float e = 0.0f;
    if (i != j && d < CUTOFF_F)
        e = 0.5f * (__cosf(d * (3.14159265358979f / CUTOFF_F)) + 1.0f);
    envm[t] = e;
}

// Pre-swizzle all GEMM B matrices into WMMA B-fragment layout (32B per lane per frag).
// B element for lane L, elem e: K = kc*32 + (e<8?e:e+8) + (L<16?0:8), N = nt*16 + L%16.
__global__ void k_swizzle(const float* __restrict__ fw1, const float* __restrict__ fw2,
                          const float* __restrict__ d1w, const float* __restrict__ d2w,
                          _Float16* __restrict__ B1, _Float16* __restrict__ B2,
                          _Float16* __restrict__ Bd1, _Float16* __restrict__ Bd2) {
    int t = blockIdx.x * blockDim.x + threadIdx.x;
    if (t >= 336 * 32) return;                                 // 48 + 3*96 frags
    int lane = t & 31;
    int frag = t >> 5;
    const float* W; _Float16* dst; int Kfull, nt, kc;
    if (frag < 48) {                                           // fw1: 3 x 8nt x 2kc
        int l = frag / 16, r = frag % 16;
        nt = r >> 1; kc = r & 1;
        W = fw1 + (size_t)l * NUM_RBF * FILTERS; Kfull = NUM_RBF;
        dst = B1 + (size_t)frag * 512;
    } else if (frag < 144) {                                   // fw2: 3 x 8nt x 4kc
        int f2 = frag - 48; int l = f2 / 32, r = f2 % 32;
        nt = r >> 2; kc = r & 3;
        W = fw2 + (size_t)l * FILTERS * FILTERS; Kfull = FILTERS;
        dst = B2 + (size_t)f2 * 512;
    } else if (frag < 240) {                                   // d1w
        int f2 = frag - 144; int l = f2 / 32, r = f2 % 32;
        nt = r >> 2; kc = r & 3;
        W = d1w + (size_t)l * HIDDEN * FILTERS; Kfull = HIDDEN;
        dst = Bd1 + (size_t)f2 * 512;
    } else {                                                   // d2w
        int f2 = frag - 240; int l = f2 / 32, r = f2 % 32;
        nt = r >> 2; kc = r & 3;
        W = d2w + (size_t)l * FILTERS * HIDDEN; Kfull = FILTERS;
        dst = Bd2 + (size_t)f2 * 512;
    }
    int n    = nt * 16 + (lane & 15);
    int loff = (lane < 16) ? 0 : 8;
    v16h v;
#pragma unroll
    for (int e = 0; e < 16; ++e) {
        int kg = kc * 32 + (e < 8 ? e : e + 8) + loff;
        float wv = (kg < Kfull) ? W[(size_t)kg * 128 + n] : 0.0f;
        v[e] = (_Float16)wv;
    }
    *(v16h*)(dst + lane * 16) = v;
}

__global__ void k_zero(float* __restrict__ p, int n) {
    int t = blockIdx.x * blockDim.x + threadIdx.x;
    if (t < n) p[t] = 0.0f;
}

// ---------------- per-layer small GEMMs (WMMA) ----------------

// xj = x @ d1w[l] + d1b[l]      (512x128 @ 128x128)
__global__ void k_gemm_xj(const _Float16* __restrict__ xf16,
                          const _Float16* __restrict__ Bd1,
                          const float* __restrict__ bias,
                          float* __restrict__ xjout) {
    const int tid = threadIdx.x;
    const int w = tid >> 5, lane = tid & 31;
    const int gw = blockIdx.x * 4 + w;                          // 256 waves
    const int mt = gw >> 3, nt = gw & 7;
    const int lcol = lane & 15, off = (lane < 16) ? 0 : 8;
    const _Float16* arow = xf16 + (size_t)(mt * 16 + lcol) * 128;
    v8f acc = {};
#pragma unroll
    for (int kc = 0; kc < 4; ++kc) {
        v16h a = load_a_f16(arow, kc, off);
        const v16h* bp = (const v16h*)(Bd1 + (size_t)(nt * 4 + kc) * 512 + lane * 16);
        acc = wmma16(a, *bp, acc);
    }
    const float bv = bias[nt * 16 + lcol];
#pragma unroll
    for (int r = 0; r < 8; ++r)
        xjout[(size_t)(mt * 16 + r + off) * 128 + nt * 16 + lcol] = acc[r] + bv;
}

// x += agg @ d2w[l] + d2b[l];  refresh xf16
__global__ void k_update(float* __restrict__ x, _Float16* __restrict__ xf16,
                         const float* __restrict__ agg,
                         const _Float16* __restrict__ Bd2,
                         const float* __restrict__ bias) {
    const int tid = threadIdx.x;
    const int w = tid >> 5, lane = tid & 31;
    const int gw = blockIdx.x * 4 + w;
    const int mt = gw >> 3, nt = gw & 7;
    const int lcol = lane & 15, off = (lane < 16) ? 0 : 8;
    const float* arow = agg + (size_t)(mt * 16 + lcol) * 128;
    v8f acc = {};
#pragma unroll
    for (int kc = 0; kc < 4; ++kc) {
        v16h a = load_a_f32(arow, kc, off);
        const v16h* bp = (const v16h*)(Bd2 + (size_t)(nt * 4 + kc) * 512 + lane * 16);
        acc = wmma16(a, *bp, acc);
    }
    const float bv = bias[nt * 16 + lcol];
#pragma unroll
    for (int r = 0; r < 8; ++r) {
        size_t idx = (size_t)(mt * 16 + r + off) * 128 + nt * 16 + lcol;
        float nv = x[idx] + acc[r] + bv;
        x[idx] = nv;
        xf16[idx] = (_Float16)nv;
    }
}

// ---------------- fused edge pipeline ----------------
// Block: 128 threads = 4 waves. blockIdx: jt = b%32 (16 j's), ig = b/32 (64 i's).
// Wave w owns i in [ig*64 + w*16, +16). Per i: one 16-edge row-tile (M = j_local).
// rbf(16x64,f16) -> GEMM1 -> +fb1 -> silu -> LDS -> GEMM2 -> +fb2 -> *env*xj -> agg.
// All weight B-fragments (48KB/layer) are staged once into LDS: the inner loop
// feeds WMMA exclusively from LDS (ds_load_b128) + registers.
__global__ void k_edge(const float* __restrict__ dmat, const float* __restrict__ envm,
                       const float* __restrict__ xj,
                       const _Float16* __restrict__ B1, const _Float16* __restrict__ B2,
                       const float* __restrict__ fb1, const float* __restrict__ fb2,
                       float* __restrict__ agg) {
    __shared__ _Float16 bfrag[48 * 512];    // B1: frags 0..15, B2: frags 16..47 (48KB)
    __shared__ _Float16 hbuf[4][16][128];   // per-wave GEMM1 output (f16)
    __shared__ float    red[16][128];       // cross-wave reduction tile

    const int tid  = threadIdx.x;
    const int w    = tid >> 5, lane = tid & 31;
    const int jt   = blockIdx.x & 31;
    const int ig   = blockIdx.x >> 5;
    const int lcol = lane & 15;
    const int off  = (lane < 16) ? 0 : 8;
    const float step = CUTOFF_F / (float)(NUM_RBF - 1);

    // Stage weight fragments to LDS (cooperative, 16B per transfer).
    {
        uint4* dst = (uint4*)bfrag;
        const uint4* s1 = (const uint4*)B1;   // 16KB = 1024 x uint4
        const uint4* s2 = (const uint4*)B2;   // 32KB = 2048 x uint4
#pragma unroll
        for (int p = 0; p < 8; ++p)  dst[tid + p * 128] = s1[tid + p * 128];
#pragma unroll
        for (int p = 0; p < 16; ++p) dst[1024 + tid + p * 128] = s2[tid + p * 128];
    }
#pragma unroll
    for (int e = 0; e < 16; ++e) red[e][tid] = 0.0f;

    const _Float16* lB1 = bfrag;
    const _Float16* lB2 = bfrag + 16 * 512;

    v8f aggacc[8];
#pragma unroll
    for (int n = 0; n < 8; ++n) aggacc[n] = (v8f){};

    __syncthreads();

    for (int ii = 0; ii < 16; ++ii) {
        const int i = ig * 64 + w * 16 + ii;
        const float dv = dmat[(size_t)i * 512 + jt * 16 + lcol];

        // Build RBF A-fragments in registers (K padded 50 -> 64).
        v16h a1[2];
#pragma unroll
        for (int kc = 0; kc < 2; ++kc) {
#pragma unroll
            for (int e = 0; e < 16; ++e) {
                int K = kc * 32 + (e < 8 ? e : e + 8) + off;
                float r = 0.0f;
                if (K < NUM_RBF) {
                    float tdiff = dv - (float)K * step;
                    r = __expf(-12.5f * tdiff * tdiff);       // 1/(2*0.2^2) = 12.5
                }
                a1[kc][e] = (_Float16)r;
            }
        }

        // GEMM1: [16x64] x [64x128] -> silu -> LDS (f16)
#pragma unroll
        for (int nt = 0; nt < 8; ++nt) {
            v8f acc = {};
            const v16h* b0 = (const v16h*)(lB1 + (size_t)(nt * 2 + 0) * 512 + lane * 16);
            const v16h* b1 = (const v16h*)(lB1 + (size_t)(nt * 2 + 1) * 512 + lane * 16);
            acc = wmma16(a1[0], *b0, acc);
            acc = wmma16(a1[1], *b1, acc);
            const float bv = fb1[nt * 16 + lcol];
#pragma unroll
            for (int r = 0; r < 8; ++r)
                hbuf[w][r + off][nt * 16 + lcol] = (_Float16)silu_f(acc[r] + bv);
        }
        __syncthreads();

        // Reload as A-fragments for GEMM2 (K = 128)
        const _Float16* hrow = &hbuf[w][lcol][0];
        v16h a2[4];
#pragma unroll
        for (int kc = 0; kc < 4; ++kc) a2[kc] = load_a_f16(hrow, kc, off);

        float envv[8];
#pragma unroll
        for (int r = 0; r < 8; ++r)
            envv[r] = envm[(size_t)i * 512 + jt * 16 + r + off];

        // GEMM2: [16x128] x [128x128]; scale by env(edge)*xj(i,f); accumulate over i
#pragma unroll
        for (int nt = 0; nt < 8; ++nt) {
            v8f acc = {};
#pragma unroll
            for (int kc = 0; kc < 4; ++kc) {
                const v16h* bp = (const v16h*)(lB2 + (size_t)(nt * 4 + kc) * 512 + lane * 16);
                acc = wmma16(a2[kc], *bp, acc);
            }
            const float bv  = fb2[nt * 16 + lcol];
            const float xjv = xj[(size_t)i * 128 + nt * 16 + lcol];
#pragma unroll
            for (int r = 0; r < 8; ++r)
                aggacc[nt][r] += (acc[r] + bv) * envv[r] * xjv;
        }
        __syncthreads();
    }

    // cross-wave reduce in LDS, then one global atomic add per element
#pragma unroll
    for (int nt = 0; nt < 8; ++nt)
#pragma unroll
        for (int r = 0; r < 8; ++r)
            atomicAdd(&red[r + off][nt * 16 + lcol], aggacc[nt][r]);
    __syncthreads();

#pragma unroll
    for (int e = 0; e < 16; ++e)
        atomicAdd(&agg[(size_t)(jt * 16 + e) * 128 + tid], red[e][tid]);
}

// ---------------- pooling + MLP head ----------------

__global__ void k_head(const float* __restrict__ x, const int* __restrict__ batch,
                       const float* __restrict__ ow1, const float* __restrict__ ob1,
                       const float* __restrict__ ow2, const float* __restrict__ ob2,
                       float* __restrict__ out) {
    __shared__ float mol[16][128];
    __shared__ float h1[16][64];
    __shared__ int   bsh[512];
    const int t = threadIdx.x;                                  // 128 threads
#pragma unroll
    for (int k = 0; k < 4; ++k) bsh[t + k * 128] = batch[t + k * 128];
    __syncthreads();

    for (int m = 0; m < N_MOL; ++m) {
        float s = 0.0f, c = 0.0f;
        for (int a = 0; a < N_ATOMS; ++a) {
            if (bsh[a] == m) { s += x[(size_t)a * 128 + t]; c += 1.0f; }
        }
        mol[m][t] = s / fmaxf(c, 1.0f);
    }
    __syncthreads();

    for (int p = t; p < 16 * 64; p += 128) {
        int m = p >> 6, k = p & 63;
        float s = ob1[k];
        for (int f = 0; f < 128; ++f) s += mol[m][f] * ow1[f * 64 + k];
        h1[m][k] = silu_f(s);
    }
    __syncthreads();

    if (t < N_MOL) {
        float s = ob2[0];
        for (int k = 0; k < 64; ++k) s += h1[t][k] * ow2[k];
        out[t] = s;
    }
}

// ---------------- host launcher ----------------

extern "C" void kernel_launch(void* const* d_in, const int* in_sizes, int n_in,
                              void* d_out, int out_size, void* d_ws, size_t ws_size,
                              hipStream_t stream) {
    (void)in_sizes; (void)n_in; (void)out_size; (void)ws_size;
    const int*   z    = (const int*)  d_in[0];
    const float* pos  = (const float*)d_in[1];
    const int*   batch= (const int*)  d_in[2];
    const float* emb  = (const float*)d_in[3];
    const float* fw1  = (const float*)d_in[4];
    const float* fb1  = (const float*)d_in[5];
    const float* fw2  = (const float*)d_in[6];
    const float* fb2  = (const float*)d_in[7];
    const float* d1w  = (const float*)d_in[8];
    const float* d1b  = (const float*)d_in[9];
    const float* d2w  = (const float*)d_in[10];
    const float* d2b  = (const float*)d_in[11];
    const float* ow1  = (const float*)d_in[12];
    const float* ob1  = (const float*)d_in[13];
    const float* ow2  = (const float*)d_in[14];
    const float* ob2  = (const float*)d_in[15];

    char* ws = (char*)d_ws;
    size_t off = 0;
    auto carve = [&](size_t bytes) { void* p = ws + off; off += (bytes + 255) & ~(size_t)255; return p; };

    float*    x    = (float*)   carve(512 * 128 * 4);
    _Float16* xf16 = (_Float16*)carve(512 * 128 * 2);
    float*    dmat = (float*)   carve(512 * 512 * 4);
    float*    envm = (float*)   carve(512 * 512 * 4);
    float*    xj   = (float*)   carve(512 * 128 * 4);
    float*    agg  = (float*)   carve(512 * 128 * 4);
    _Float16* B1   = (_Float16*)carve((size_t)48  * 1024);     // 3 x 16 frags x 1KB
    _Float16* B2   = (_Float16*)carve((size_t)96  * 1024);     // 3 x 32 frags
    _Float16* Bd1  = (_Float16*)carve((size_t)96  * 1024);
    _Float16* Bd2  = (_Float16*)carve((size_t)96  * 1024);

    k_embed   <<<256, 256, 0, stream>>>(z, emb, x, xf16);
    k_dist_env<<<1024, 256, 0, stream>>>(pos, dmat, envm);
    k_swizzle <<<42, 256, 0, stream>>>(fw1, fw2, d1w, d2w, B1, B2, Bd1, Bd2);

    for (int l = 0; l < N_LAYERS; ++l) {
        k_gemm_xj<<<64, 128, 0, stream>>>(xf16, Bd1 + (size_t)l * 32 * 512,
                                          d1b + l * 128, xj);
        k_zero<<<256, 256, 0, stream>>>(agg, 512 * 128);
        k_edge<<<256, 128, 0, stream>>>(dmat, envm, xj,
                                        B1 + (size_t)l * 16 * 512,
                                        B2 + (size_t)l * 32 * 512,
                                        fb1 + l * 128, fb2 + l * 128, agg);
        k_update<<<64, 128, 0, stream>>>(x, xf16, agg,
                                         Bd2 + (size_t)l * 32 * 512,
                                         d2b + l * 128);
    }

    k_head<<<1, 128, 0, stream>>>(x, batch, ow1, ob1, ow2, ob2, (float*)d_out);
}
